// ClefDecoder_38482906972939
// MI455X (gfx1250) — matrix operand order repaired and checked by reading, not compiled, not run on hardware
//
#include <hip/hip_runtime.h>

// ---------------- static config ----------------
#define DM     512
#define S_LEN  1024
#define NH     8
#define HDIM   64
#define BATCH  4
#define NTOT   43520
#define ROWS   (BATCH * S_LEN)        // 4096
#define FF     2048
#define NLVL   4
#define NPTS   4
#define LSTR   40                     // LDS row stride (bf16 elems): keeps 16B alignment

#ifndef __has_builtin
#define __has_builtin(x) 0
#endif
#if __has_builtin(__builtin_amdgcn_global_load_async_to_lds_b128)
#define HAVE_ASYNC_LDS 1
#else
#define HAVE_ASYNC_LDS 0
#endif

typedef __attribute__((ext_vector_type(16))) __bf16 v16bf;
typedef __attribute__((ext_vector_type(8)))  float  f32x8;

// types matching the async-to-LDS builtin signature (from hipcc diagnostic):
typedef int v4i_vs __attribute__((vector_size(16)));
typedef __attribute__((address_space(1))) v4i_vs* as1_v4i;
typedef __attribute__((address_space(3))) v4i_vs* as3_v4i;

union FragB { v16bf v; unsigned int u[8]; unsigned short s[16]; };

__device__ __forceinline__ unsigned short f2bf(float f) {
  unsigned int u = __float_as_uint(f);
  u += 0x7FFFu + ((u >> 16) & 1u);           // round-to-nearest-even
  return (unsigned short)(u >> 16);
}
__device__ __forceinline__ float bf2f(unsigned short h) {
  return __uint_as_float(((unsigned int)h) << 16);
}
__device__ __forceinline__ f32x8 wmma_bf(const FragB& a, const FragB& b, f32x8 c) {
  return __builtin_amdgcn_wmma_f32_16x16x32_bf16(false, a.v, false, b.v, (short)0, c, false, false);
}
__device__ __forceinline__ float gelu_exact(float x) {
  return 0.5f * x * (1.0f + erff(x * 0.70710678118654752f));
}
__device__ __forceinline__ float softplus_f(float x) {
  return (x > 20.0f) ? x : log1pf(__expf(x));
}
__device__ __forceinline__ float sigmoid_f(float x) { return 1.0f / (1.0f + __expf(-x)); }

// A-matrix (16xK row-contiguous) bf16 fragment k-index for slot v, half hh
__device__ __forceinline__ int a_kidx(int v, int hh) { return ((v & 3) * 2) + ((v >> 2) * 16) + hh * 8; }
// B-matrix (Kx16) bf16 fragment k-index for slot v, half hh
__device__ __forceinline__ int b_kidx(int v, int hh) { return ((v & 3) * 2) + ((v >> 2) * 16) + hh * 8; }

__device__ __forceinline__ void wait_async_lds() {
#if HAVE_ASYNC_LDS
#if __has_builtin(__builtin_amdgcn_s_wait_asynccnt)
  __builtin_amdgcn_s_wait_asynccnt(0);
#else
  asm volatile("s_wait_asynccnt 0" ::: "memory");
#endif
#endif
}

// ---------------- elementwise helpers ----------------
__global__ void cvt_bf16_kernel(const float* __restrict__ in, unsigned short* __restrict__ out, size_t n) {
  size_t i = (size_t)blockIdx.x * blockDim.x + threadIdx.x;
  size_t st = (size_t)gridDim.x * blockDim.x;
  for (; i < n; i += st) out[i] = f2bf(in[i]);
}

__global__ void add_cvt_bf16_kernel(const float* __restrict__ a, const float* __restrict__ b,
                                    unsigned short* __restrict__ out, size_t n) {
  size_t i = (size_t)blockIdx.x * blockDim.x + threadIdx.x;
  size_t st = (size_t)gridDim.x * blockDim.x;
  for (; i < n; i += st) out[i] = f2bf(a[i] + b[i]);
}

// ---------------- GEMM: C[M,N] = A_bf16[M,K] @ W_bf16[K,N] (+bias)(+gelu) ----------------
// block: 256 thr = 8 waves; tile 64(M) x 64(N); K-step 32.
// wave (wm = w&3, wn = w>>2) computes a 16x32 output strip: 2 WMMAs/step, A-frag reused.
template <int EPI>   // 0 = none, 1 = bias, 2 = bias + exact gelu
__global__ void gemm_bf16_kernel(const unsigned short* __restrict__ A,
                                 const unsigned short* __restrict__ W,
                                 const float* __restrict__ bias,
                                 float* __restrict__ outF, unsigned short* __restrict__ outH,
                                 int M, int N, int K) {
  __shared__ unsigned short lA[64 * LSTR];
  __shared__ unsigned short lB[64 * LSTR];
  const int tid  = threadIdx.x;
  const int lane = tid & 31;
  const int w    = tid >> 5;
  const int wm   = w & 3;          // 0..3  (16-row group)
  const int wn   = w >> 2;         // 0..1  (32-col group)
  const int bm   = blockIdx.y * 64;
  const int bn   = blockIdx.x * 64;
  const int r    = lane & 15;
  const int hh   = lane >> 4;

  // staging assignments (8 bf16 = 16B per thread for both tiles)
  const int ea = tid * 8;
  const int ar = ea >> 5, ac = ea & 31;     // A tile 64 x 32
  const int wr = ea >> 6, wc = ea & 63;     // W tile 32 x 64

  f32x8 acc0 = {}, acc1 = {};

  for (int k0 = 0; k0 < K; k0 += 32) {
    const unsigned short* asrc = A + (size_t)(bm + ar) * K + k0 + ac;
#if HAVE_ASYNC_LDS
    __builtin_amdgcn_global_load_async_to_lds_b128(
        (as1_v4i)(void*)asrc,
        (as3_v4i)(void*)&lA[ar * LSTR + ac], 0, 0);
#else
    {
      uint4 da = *(const uint4*)asrc;
      *(uint4*)&lA[ar * LSTR + ac] = da;
    }
#endif
    // stage W tile 32x64 transposed into lB[n][k]
    {
      const unsigned short* wsrc = W + (size_t)(k0 + wr) * N + bn + wc;
      uint4 dw = *(const uint4*)wsrc;
      unsigned short t[8];
      *(uint4*)t = dw;
#pragma unroll
      for (int i = 0; i < 8; ++i) lB[(wc + i) * LSTR + wr] = t[i];
    }
    if (k0 + 32 < K) {   // prefetch next K tile (global_prefetch_b8)
      __builtin_prefetch(asrc + 32, 0, 1);
      __builtin_prefetch(W + (size_t)(k0 + 32 + wr) * N + bn + wc, 0, 1);
    }
    wait_async_lds();
    __syncthreads();

    FragB af, b0, b1;
#pragma unroll
    for (int v = 0; v < 8; ++v) {
      af.u[v] = *(const unsigned int*)&lA[(wm * 16 + r) * LSTR + a_kidx(v, hh)];
      b0.u[v] = *(const unsigned int*)&lB[(wn * 32 + r) * LSTR + b_kidx(v, hh)];
      b1.u[v] = *(const unsigned int*)&lB[(wn * 32 + 16 + r) * LSTR + b_kidx(v, hh)];
    }
    acc0 = wmma_bf(af, b0, acc0);
    acc1 = wmma_bf(af, b1, acc1);
    __syncthreads();
  }

  const int n = lane & 15;
#pragma unroll
  for (int cg = 0; cg < 2; ++cg) {
    const int col = bn + wn * 32 + cg * 16 + n;
    const float bv = (EPI >= 1) ? bias[col] : 0.0f;
#pragma unroll
    for (int r8 = 0; r8 < 8; ++r8) {
      int row = bm + wm * 16 + r8 + 8 * hh;
      float vv = (cg ? acc1[r8] : acc0[r8]) + bv;
      if (EPI == 2) vv = gelu_exact(vv);
      size_t idx = (size_t)row * N + col;
      if (outF) outF[idx] = vv;
      if (outH) outH[idx] = f2bf(vv);
    }
  }
}

// ---------------- QKV repack ----------------
// qkv f32 [ROWS,1536] -> Q,K bf16 [B*H, S, 64]; V bf16 TRANSPOSED [B*H, 64, S]
__global__ void repack_qkv_kernel(const float* __restrict__ qkv,
                                  unsigned short* __restrict__ Qb,
                                  unsigned short* __restrict__ Kb,
                                  unsigned short* __restrict__ Vtb) {
  int i = blockIdx.x * blockDim.x + threadIdx.x;
  if (i >= ROWS * DM) return;
  int row = i >> 9, c = i & 511;
  int h = c >> 6, d = c & 63;
  int b = row >> 10, s = row & 1023;
  int bh = b * NH + h;
  size_t dst  = ((size_t)bh * S_LEN + s) * HDIM + d;
  size_t dstT = ((size_t)bh * HDIM + d) * S_LEN + s;
  const float* src = qkv + (size_t)row * (3 * DM);
  Qb[dst]   = f2bf(src[c]);
  Kb[dst]   = f2bf(src[DM + c]);
  Vtb[dstT] = f2bf(src[2 * DM + c]);
}

// ---------------- flash causal attention, WMMA bf16, one wave per 16 query rows ----------------
__global__ void attn_kernel(const unsigned short* __restrict__ Qb,
                            const unsigned short* __restrict__ Kb,
                            const unsigned short* __restrict__ Vtb,
                            unsigned short* __restrict__ sa_bf) {
  __shared__ unsigned short ldsP[8 * 16 * 34];
  const int wave = blockIdx.x * (blockDim.x >> 5) + (threadIdx.x >> 5);
  const int lane = threadIdx.x & 31;
  const int s16  = wave & (S_LEN / 16 - 1);
  const int bh   = wave >> 6;                // 0..31
  const int h    = bh & (NH - 1);
  const int b    = bh >> 3;
  const int q0   = s16 * 16;
  const int r    = lane & 15;
  const int hh   = lane >> 4;
  const int n    = lane & 15;

  const unsigned short* Qp  = Qb  + (size_t)bh * S_LEN * HDIM;
  const unsigned short* Kp  = Kb  + (size_t)bh * S_LEN * HDIM;
  const unsigned short* Vtp = Vtb + (size_t)bh * HDIM * S_LEN;
  unsigned short* lp = ldsP + (threadIdx.x >> 5) * (16 * 34);

  // Q fragments (two 32-wide chunks of head dim)
  FragB qf[2];
#pragma unroll
  for (int c = 0; c < 2; ++c)
#pragma unroll
    for (int v = 0; v < 8; ++v)
      qf[c].u[v] = *(const unsigned int*)(Qp + (size_t)(q0 + r) * HDIM + a_kidx(v, hh) + c * 32);

  f32x8 acc[4] = {};
  float mi[8], li[8];
#pragma unroll
  for (int j = 0; j < 8; ++j) { mi[j] = -1e30f; li[j] = 0.0f; }

  for (int kc = 0; kc * 32 <= q0 + 15; ++kc) {
    const int kvb0 = kc * 32;
    const int kvb1 = kvb0 + 16;
    const bool t1act = (kvb1 <= q0 + 15);

    // ---- scores ----
    f32x8 s0 = {}, s1 = {};
#pragma unroll
    for (int c = 0; c < 2; ++c) {
      FragB kf;
#pragma unroll
      for (int v = 0; v < 8; ++v)
        kf.u[v] = *(const unsigned int*)(Kp + (size_t)(kvb0 + n) * HDIM + b_kidx(v, hh) + c * 32);
      s0 = wmma_bf(qf[c], kf, s0);
    }
    if (t1act) {
#pragma unroll
      for (int c = 0; c < 2; ++c) {
        FragB kf;
#pragma unroll
        for (int v = 0; v < 8; ++v)
          kf.u[v] = *(const unsigned int*)(Kp + (size_t)(kvb1 + n) * HDIM + b_kidx(v, hh) + c * 32);
        s1 = wmma_bf(qf[c], kf, s1);
      }
    }

    // ---- scale + causal mask + online softmax ----
    float p0[8], p1[8], tmax[8];
    const int col0 = kvb0 + n, col1 = kvb1 + n;
#pragma unroll
    for (int j = 0; j < 8; ++j) {
      int m = j + 8 * hh;          // row within tile
      float a = s0[j] * 0.125f;
      if (col0 > q0 + m) a = -1e30f;
      float c2 = t1act ? s1[j] * 0.125f : -1e30f;
      if (col1 > q0 + m) c2 = -1e30f;
      p0[j] = a; p1[j] = c2;
      tmax[j] = fmaxf(a, c2);
    }
#pragma unroll
    for (int msk = 1; msk < 16; msk <<= 1)
#pragma unroll
      for (int j = 0; j < 8; ++j) tmax[j] = fmaxf(tmax[j], __shfl_xor(tmax[j], msk, 32));

    float corr[8];
#pragma unroll
    for (int j = 0; j < 8; ++j) {
      float mn = fmaxf(mi[j], tmax[j]);
      corr[j] = __expf(mi[j] - mn);
      mi[j] = mn;
      p0[j] = __expf(p0[j] - mn);
      p1[j] = t1act ? __expf(p1[j] - mn) : 0.0f;
    }
    float rs[8];
#pragma unroll
    for (int j = 0; j < 8; ++j) rs[j] = p0[j] + p1[j];
#pragma unroll
    for (int msk = 1; msk < 16; msk <<= 1)
#pragma unroll
      for (int j = 0; j < 8; ++j) rs[j] += __shfl_xor(rs[j], msk, 32);
#pragma unroll
    for (int j = 0; j < 8; ++j) li[j] = li[j] * corr[j] + rs[j];
#pragma unroll
    for (int g = 0; g < 4; ++g)
#pragma unroll
      for (int j = 0; j < 8; ++j) acc[g][j] *= corr[j];

    // ---- P (D layout) -> LDS (row-major 16x32) ----
    asm volatile("s_wait_dscnt 0" ::: "memory");   // prior-frag reads done before overwrite
#pragma unroll
    for (int j = 0; j < 8; ++j) {
      int m = j + 8 * hh;
      lp[m * 34 + n]      = f2bf(p0[j]);
      lp[m * 34 + 16 + n] = f2bf(p1[j]);
    }
    asm volatile("s_wait_dscnt 0" ::: "memory");

    FragB pf;
#pragma unroll
    for (int v = 0; v < 8; ++v)
      pf.u[v] = *(const unsigned int*)&lp[(lane & 15) * 34 + a_kidx(v, hh)];

    // ---- PV: V transposed [hd][S] -> contiguous 32-bit K-pairs ----
#pragma unroll
    for (int g = 0; g < 4; ++g) {
      FragB vf;
#pragma unroll
      for (int v = 0; v < 8; ++v)
        vf.u[v] = *(const unsigned int*)(Vtp + (size_t)(g * 16 + n) * S_LEN + kvb0 + b_kidx(v, hh));
      acc[g] = wmma_bf(pf, vf, acc[g]);
    }
  }

  // ---- epilogue: normalize, write bf16 [row, h*64 + col] ----
#pragma unroll
  for (int g = 0; g < 4; ++g)
#pragma unroll
    for (int j = 0; j < 8; ++j) {
      int m = j + 8 * hh;
      float o = acc[g][j] / li[j];
      sa_bf[((size_t)(b * S_LEN + q0 + m)) * DM + h * HDIM + g * 16 + n] = f2bf(o);
    }
}

// ---------------- LayerNorm (wave per row) ----------------
__global__ void ln_residual_kernel(const float* __restrict__ base, const float* __restrict__ add,
                                   const float* __restrict__ g, const float* __restrict__ bta,
                                   float* __restrict__ outF, unsigned short* __restrict__ outH,
                                   int nrows) {
  int wid  = blockIdx.x * (blockDim.x >> 5) + (threadIdx.x >> 5);
  int lane = threadIdx.x & 31;
  if (wid >= nrows) return;
  const float* pb = base + (size_t)wid * DM;
  const float* pa = add  + (size_t)wid * DM;
  float u[16]; float s = 0.0f;
#pragma unroll
  for (int j = 0; j < 16; ++j) { int c = j * 32 + lane; u[j] = pb[c] + pa[c]; s += u[j]; }
#pragma unroll
  for (int m = 1; m < 32; m <<= 1) s += __shfl_xor(s, m, 32);
  float mean = s * (1.0f / DM);
  float vs = 0.0f;
#pragma unroll
  for (int j = 0; j < 16; ++j) { float d = u[j] - mean; vs += d * d; }
#pragma unroll
  for (int m = 1; m < 32; m <<= 1) vs += __shfl_xor(vs, m, 32);
  float inv = rsqrtf(vs * (1.0f / DM) + 1e-5f);
#pragma unroll
  for (int j = 0; j < 16; ++j) {
    int c = j * 32 + lane;
    float o = (u[j] - mean) * inv * g[c] + bta[c];
    if (outF) outF[(size_t)wid * DM + c] = o;
    if (outH) outH[(size_t)wid * DM + c] = f2bf(o);
  }
}

// PC gate + LN2: u = base + softplus(dot(base,gw)+gb) * (tgt2 - pred); LN(u)
__global__ void ln_gate_kernel(const float* __restrict__ base, const float* __restrict__ tgt2,
                               const float* __restrict__ pred, const float* __restrict__ gw,
                               const float* __restrict__ gb, const float* __restrict__ g,
                               const float* __restrict__ bta,
                               float* __restrict__ outF, unsigned short* __restrict__ outH,
                               int nrows) {
  int wid  = blockIdx.x * (blockDim.x >> 5) + (threadIdx.x >> 5);
  int lane = threadIdx.x & 31;
  if (wid >= nrows) return;
  const float* pb = base + (size_t)wid * DM;
  const float* p2 = tgt2 + (size_t)wid * DM;
  const float* pp = pred + (size_t)wid * DM;
  float bx[16]; float dg = 0.0f;
#pragma unroll
  for (int j = 0; j < 16; ++j) { int c = j * 32 + lane; bx[j] = pb[c]; dg += bx[j] * gw[c]; }
#pragma unroll
  for (int m = 1; m < 32; m <<= 1) dg += __shfl_xor(dg, m, 32);
  float gain = softplus_f(dg + gb[0]);
  float u[16]; float s = 0.0f;
#pragma unroll
  for (int j = 0; j < 16; ++j) { int c = j * 32 + lane; u[j] = bx[j] + gain * (p2[c] - pp[c]); s += u[j]; }
#pragma unroll
  for (int m = 1; m < 32; m <<= 1) s += __shfl_xor(s, m, 32);
  float mean = s * (1.0f / DM);
  float vs = 0.0f;
#pragma unroll
  for (int j = 0; j < 16; ++j) { float d = u[j] - mean; vs += d * d; }
#pragma unroll
  for (int m = 1; m < 32; m <<= 1) vs += __shfl_xor(vs, m, 32);
  float inv = rsqrtf(vs * (1.0f / DM) + 1e-5f);
#pragma unroll
  for (int j = 0; j < 16; ++j) {
    int c = j * 32 + lane;
    float o = (u[j] - mean) * inv * g[c] + bta[c];
    if (outF) outF[(size_t)wid * DM + c] = o;
    if (outH) outH[(size_t)wid * DM + c] = f2bf(o);
  }
}

// ---------------- reference points (wave per row) ----------------
__global__ void ref_kernel(const float* __restrict__ tgt_pos, const float* __restrict__ ln1,
                           const float* __restrict__ timeW, const float* __restrict__ timeB,
                           const float* __restrict__ freqW, const float* __restrict__ freqB,
                           const float* __restrict__ refW, const float* __restrict__ refB,
                           float* __restrict__ refOut, int nrows) {
  int wid  = blockIdx.x * (blockDim.x >> 5) + (threadIdx.x >> 5);
  int lane = threadIdx.x & 31;
  if (wid >= nrows) return;
  const float* tp = tgt_pos + (size_t)wid * DM;
  const float* lv = ln1     + (size_t)wid * DM;
  float st = 0, sf = 0, s0 = 0, s1 = 0;
#pragma unroll
  for (int j = 0; j < 16; ++j) {
    int c = j * 32 + lane;
    float a = tp[c], b = lv[c];
    st += a * timeW[c];
    sf += b * freqW[c];
    s0 += b * refW[c * 2];
    s1 += b * refW[c * 2 + 1];
  }
#pragma unroll
  for (int m = 1; m < 32; m <<= 1) {
    st += __shfl_xor(st, m, 32); sf += __shfl_xor(sf, m, 32);
    s0 += __shfl_xor(s0, m, 32); s1 += __shfl_xor(s1, m, 32);
  }
  float t  = sigmoid_f(st + timeB[0]);
  float f  = sigmoid_f(sf + freqB[0]);
  float rx = tanhf(s0 + refB[0]) * 0.1f;
  float ry = tanhf(s1 + refB[1]) * 0.1f;
  float rt = fminf(fmaxf(t + rx, 0.0f), 1.0f);
  float rf = fminf(fmaxf(f + ry, 0.0f), 1.0f);
  if (lane == 0) {
#pragma unroll
    for (int h = 0; h < NH; ++h) {
      refOut[((size_t)wid * NH + h) * 2 + 0] = rt;
      refOut[((size_t)wid * NH + h) * 2 + 1] = rf;
    }
  }
}

// ---------------- attention-weight softmax over 16 ----------------
__global__ void aw_softmax_kernel(const float* __restrict__ in, float* __restrict__ out) {
  int i = blockIdx.x * blockDim.x + threadIdx.x;
  if (i >= ROWS * NH) return;
  const float* p = in + (size_t)i * 16;
  float mx = -1e30f;
#pragma unroll
  for (int j = 0; j < 16; ++j) mx = fmaxf(mx, p[j]);
  float e[16]; float s = 0;
#pragma unroll
  for (int j = 0; j < 16; ++j) { e[j] = __expf(p[j] - mx); s += e[j]; }
  float inv = 1.0f / s;
  float* o = out + (size_t)i * 16;
#pragma unroll
  for (int j = 0; j < 16; ++j) o[j] = e[j] * inv;
}

// ---------------- deformable bilinear sampling (wave per (b,s,h); lane = 2 dims) ----------------
__global__ void flux_sample_kernel(const unsigned short* __restrict__ valb,
                                   const float* __restrict__ ref,
                                   const float* __restrict__ off,
                                   const float* __restrict__ aw,
                                   const float* __restrict__ vr,
                                   unsigned short* __restrict__ outb) {
  const int LH[4] = {128, 64, 32, 16};
  const int LW[4] = {256, 128, 64, 32};
  const int LS[4] = {0, 32768, 40960, 43008};
  int wid  = blockIdx.x * (blockDim.x >> 5) + (threadIdx.x >> 5);
  int lane = threadIdx.x & 31;
  if (wid >= ROWS * NH) return;
  int h = wid & (NH - 1);
  int row = wid >> 3;
  int b = row >> 10;
  float rt = ref[((size_t)row * NH + h) * 2 + 0];
  float rf = ref[((size_t)row * NH + h) * 2 + 1];
  const float* offp = off + (size_t)row * (NH * NLVL * NPTS * 2) + h * (NLVL * NPTS * 2);
  const float* awp  = aw  + (size_t)row * (NH * NLVL * NPTS)    + h * (NLVL * NPTS);
  const unsigned short* vbase = valb + (size_t)b * NTOT * DM + h * HDIM + lane * 2;
  float a0 = 0.0f, a1 = 0.0f;
#pragma unroll
  for (int l = 0; l < NLVL; ++l) {
    const int Hl = LH[l], Wl = LW[l], st = LS[l];
    float vrx = vr[(b * NLVL + l) * 2 + 0];
    float vry = vr[(b * NLVL + l) * 2 + 1];
#pragma unroll
    for (int p = 0; p < NPTS; ++p) {
      float ox = offp[l * (NPTS * 2) + p * 2 + 0];
      float oy = offp[l * (NPTS * 2) + p * 2 + 1];
      float lx = rt * vrx + ox * (1.0f / (float)Wl);
      float ly = rf * vry + oy * (1.0f / (float)Hl);
      float x = lx * (float)Wl - 0.5f;
      float y = ly * (float)Hl - 0.5f;
      float x0 = floorf(x), y0 = floorf(y);
      float dx = x - x0, dy = y - y0;
      float aww = awp[l * NPTS + p];
#pragma unroll
      for (int cc = 0; cc < 4; ++cc) {
        float cx = x0 + (float)(cc & 1);
        float cy = y0 + (float)(cc >> 1);
        float wx = (cc & 1) ? dx : (1.0f - dx);
        float wy = (cc >> 1) ? dy : (1.0f - dy);
        bool inb = (cx >= 0.0f) && (cx < (float)Wl) && (cy >= 0.0f) && (cy < (float)Hl);
        if (inb) {
          int xi = (int)cx, yi = (int)cy;
          size_t idx = (size_t)(st + yi * Wl + xi) * DM;
          unsigned int pk = *(const unsigned int*)(vbase + idx);
          float wgt = wx * wy * aww;
          a0 += wgt * bf2f((unsigned short)(pk & 0xFFFFu));
          a1 += wgt * bf2f((unsigned short)(pk >> 16));
        }
      }
    }
  }
  unsigned int pk = (unsigned int)f2bf(a0) | ((unsigned int)f2bf(a1) << 16);
  *(unsigned int*)(outb + (size_t)row * DM + h * HDIM + lane * 2) = pk;
}

// ---------------- launcher ----------------
extern "C" void kernel_launch(void* const* d_in, const int* in_sizes, int n_in,
                              void* d_out, int out_size, void* d_ws, size_t ws_size,
                              hipStream_t stream) {
  (void)in_sizes; (void)n_in; (void)out_size; (void)ws_size;
  const float* tgt     = (const float*)d_in[0];
  const float* memory  = (const float*)d_in[1];
  const float* vratios = (const float*)d_in[4];
  const float* tgt_pos = (const float*)d_in[5];
  const float* qkv_w = (const float*)d_in[6];   const float* qkv_b = (const float*)d_in[7];
  const float* so_w  = (const float*)d_in[8];   const float* so_b  = (const float*)d_in[9];
  const float* n1g   = (const float*)d_in[10];  const float* n1b   = (const float*)d_in[11];
  const float* n2g   = (const float*)d_in[12];  const float* n2b   = (const float*)d_in[13];
  const float* n3g   = (const float*)d_in[14];  const float* n3b   = (const float*)d_in[15];
  const float* val_w = (const float*)d_in[16];  const float* val_b = (const float*)d_in[17];
  const float* off_w = (const float*)d_in[18];  const float* off_b = (const float*)d_in[19];
  const float* aw_w  = (const float*)d_in[20];  const float* aw_b  = (const float*)d_in[21];
  const float* fo_w  = (const float*)d_in[22];  const float* fo_b  = (const float*)d_in[23];
  const float* p1_w  = (const float*)d_in[24];  const float* p1_b  = (const float*)d_in[25];
  const float* p2_w  = (const float*)d_in[26];  const float* p2_b  = (const float*)d_in[27];
  const float* gn_w  = (const float*)d_in[28];  const float* gn_b  = (const float*)d_in[29];
  const float* f1_w  = (const float*)d_in[30];  const float* f1_b  = (const float*)d_in[31];
  const float* f2_w  = (const float*)d_in[32];  const float* f2_b  = (const float*)d_in[33];
  const float* tm_w  = (const float*)d_in[34];  const float* tm_b  = (const float*)d_in[35];
  const float* fq_w  = (const float*)d_in[36];  const float* fq_b  = (const float*)d_in[37];
  const float* rf_w  = (const float*)d_in[38];  const float* rf_b  = (const float*)d_in[39];

  char* ws = (char*)d_ws; size_t cur = 0;
  auto carve = [&](size_t bytes) -> void* {
    void* p = ws + cur; cur = (cur + bytes + 255) & ~(size_t)255; return p;
  };
  unsigned short* wb_qkv = (unsigned short*)carve((size_t)DM * 3 * DM * 2);
  unsigned short* wb_so  = (unsigned short*)carve((size_t)DM * DM * 2);
  unsigned short* wb_val = (unsigned short*)carve((size_t)DM * DM * 2);
  unsigned short* wb_off = (unsigned short*)carve((size_t)DM * 256 * 2);
  unsigned short* wb_aw  = (unsigned short*)carve((size_t)DM * 128 * 2);
  unsigned short* wb_fo  = (unsigned short*)carve((size_t)DM * DM * 2);
  unsigned short* wb_p1  = (unsigned short*)carve((size_t)DM * DM * 2);
  unsigned short* wb_p2  = (unsigned short*)carve((size_t)DM * DM * 2);
  unsigned short* wb_f1  = (unsigned short*)carve((size_t)DM * FF * 2);
  unsigned short* wb_f2  = (unsigned short*)carve((size_t)FF * DM * 2);
  unsigned short* mem_bf = (unsigned short*)carve((size_t)BATCH * NTOT * DM * 2);
  unsigned short* val_bf = (unsigned short*)carve((size_t)BATCH * NTOT * DM * 2);
  unsigned short* x_bf   = (unsigned short*)carve((size_t)ROWS * DM * 2);
  float*          qkv_f  = (float*)carve((size_t)ROWS * 3 * DM * 4);
  unsigned short* q_bf   = (unsigned short*)carve((size_t)ROWS * DM * 2);
  unsigned short* k_bf   = (unsigned short*)carve((size_t)ROWS * DM * 2);
  unsigned short* vt_bf  = (unsigned short*)carve((size_t)ROWS * DM * 2);
  unsigned short* sa_bf  = (unsigned short*)carve((size_t)ROWS * DM * 2);
  float*          saY_f  = (float*)carve((size_t)ROWS * DM * 4);
  float*          ln1_f  = (float*)carve((size_t)ROWS * DM * 4);
  unsigned short* ln1_bf = (unsigned short*)carve((size_t)ROWS * DM * 2);
  float*          ref_f  = (float*)carve((size_t)ROWS * NH * 2 * 4);
  float*          off_f  = (float*)carve((size_t)ROWS * 256 * 4);
  float*          aw_f   = (float*)carve((size_t)ROWS * 128 * 4);
  float*          aw_s   = (float*)carve((size_t)ROWS * 128 * 4);
  unsigned short* flux_bf= (unsigned short*)carve((size_t)ROWS * DM * 2);
  float*          tgt2_f = (float*)carve((size_t)ROWS * DM * 4);
  unsigned short* p1_bf  = (unsigned short*)carve((size_t)ROWS * DM * 2);
  float*          pr2_f  = (float*)carve((size_t)ROWS * DM * 4);
  float*          ln2_f  = (float*)carve((size_t)ROWS * DM * 4);
  unsigned short* ln2_bf = (unsigned short*)carve((size_t)ROWS * DM * 2);
  unsigned short* h1_bf  = (unsigned short*)carve((size_t)ROWS * FF * 2);
  float*          y2_f   = (float*)carve((size_t)ROWS * DM * 4);

  const dim3 blk(256);
  auto cvt = [&](const float* src, unsigned short* dst, size_t n) {
    int g = (int)((n + 255) / 256); if (g > 4096) g = 4096;
    cvt_bf16_kernel<<<dim3(g), blk, 0, stream>>>(src, dst, n);
  };
  // 1) weight + memory conversions to bf16
  cvt(qkv_w, wb_qkv, (size_t)DM * 3 * DM);
  cvt(so_w,  wb_so,  (size_t)DM * DM);
  cvt(val_w, wb_val, (size_t)DM * DM);
  cvt(off_w, wb_off, (size_t)DM * 256);
  cvt(aw_w,  wb_aw,  (size_t)DM * 128);
  cvt(fo_w,  wb_fo,  (size_t)DM * DM);
  cvt(p1_w,  wb_p1,  (size_t)DM * DM);
  cvt(p2_w,  wb_p2,  (size_t)DM * DM);
  cvt(f1_w,  wb_f1,  (size_t)DM * FF);
  cvt(f2_w,  wb_f2,  (size_t)FF * DM);
  cvt(memory, mem_bf, (size_t)BATCH * NTOT * DM);

  // 2) x = bf16(tgt + tgt_pos)
  add_cvt_bf16_kernel<<<dim3(4096), blk, 0, stream>>>(tgt, tgt_pos, x_bf, (size_t)ROWS * DM);

  // 3) QKV GEMM (+bias) -> fp32, then repack (V transposed for PV fragments)
  gemm_bf16_kernel<1><<<dim3(3 * DM / 64, ROWS / 64), blk, 0, stream>>>(
      x_bf, wb_qkv, qkv_b, qkv_f, (unsigned short*)nullptr, ROWS, 3 * DM, DM);
  repack_qkv_kernel<<<dim3(ROWS * DM / 256), blk, 0, stream>>>(qkv_f, q_bf, k_bf, vt_bf);

  // 4) flash causal attention -> sa_bf [ROWS, DM]
  attn_kernel<<<dim3(BATCH * NH * (S_LEN / 16) / 8), blk, 0, stream>>>(q_bf, k_bf, vt_bf, sa_bf);

  // 5) sa_out GEMM + residual LN1
  gemm_bf16_kernel<1><<<dim3(DM / 64, ROWS / 64), blk, 0, stream>>>(
      sa_bf, wb_so, so_b, saY_f, (unsigned short*)nullptr, ROWS, DM, DM);
  ln_residual_kernel<<<dim3(ROWS / 8), blk, 0, stream>>>(tgt, saY_f, n1g, n1b, ln1_f, ln1_bf, ROWS);

  // 6) reference points
  ref_kernel<<<dim3(ROWS / 8), blk, 0, stream>>>(tgt_pos, ln1_f, tm_w, tm_b, fq_w, fq_b,
                                                 rf_w, rf_b, ref_f, ROWS);

  // 7) value projection (big GEMM) -> bf16 (178MB: stays L2-resident for sampling)
  gemm_bf16_kernel<1><<<dim3(DM / 64, BATCH * NTOT / 64), blk, 0, stream>>>(
      mem_bf, wb_val, val_b, (float*)nullptr, val_bf, BATCH * NTOT, DM, DM);

  // 8) offsets / attention weights
  gemm_bf16_kernel<1><<<dim3(256 / 64, ROWS / 64), blk, 0, stream>>>(
      ln1_bf, wb_off, off_b, off_f, (unsigned short*)nullptr, ROWS, 256, DM);
  gemm_bf16_kernel<1><<<dim3(128 / 64, ROWS / 64), blk, 0, stream>>>(
      ln1_bf, wb_aw, aw_b, aw_f, (unsigned short*)nullptr, ROWS, 128, DM);
  aw_softmax_kernel<<<dim3(ROWS * NH / 256), blk, 0, stream>>>(aw_f, aw_s);

  // 9) deformable sampling -> flux input (bf16)
  flux_sample_kernel<<<dim3(ROWS * NH / 8), blk, 0, stream>>>(val_bf, ref_f, off_f, aw_s,
                                                              vratios, flux_bf);

  // 10) flux output projection
  gemm_bf16_kernel<1><<<dim3(DM / 64, ROWS / 64), blk, 0, stream>>>(
      flux_bf, wb_fo, fo_b, tgt2_f, (unsigned short*)nullptr, ROWS, DM, DM);

  // 11) predictor: gelu(ln1 @ p1) @ p2
  gemm_bf16_kernel<2><<<dim3(DM / 64, ROWS / 64), blk, 0, stream>>>(
      ln1_bf, wb_p1, p1_b, (float*)nullptr, p1_bf, ROWS, DM, DM);
  gemm_bf16_kernel<1><<<dim3(DM / 64, ROWS / 64), blk, 0, stream>>>(
      p1_bf, wb_p2, p2_b, pr2_f, (unsigned short*)nullptr, ROWS, DM, DM);

  // 12) PC gate + LN2
  ln_gate_kernel<<<dim3(ROWS / 8), blk, 0, stream>>>(ln1_f, tgt2_f, pr2_f, gn_w, gn_b,
                                                     n2g, n2b, ln2_f, ln2_bf, ROWS);

  // 13) FFN + final LN -> d_out
  gemm_bf16_kernel<2><<<dim3(FF / 64, ROWS / 64), blk, 0, stream>>>(
      ln2_bf, wb_f1, f1_b, (float*)nullptr, h1_bf, ROWS, FF, DM);
  gemm_bf16_kernel<1><<<dim3(DM / 64, ROWS / 64), blk, 0, stream>>>(
      h1_bf, wb_f2, f2_b, y2_f, (unsigned short*)nullptr, ROWS, DM, FF);
  ln_residual_kernel<<<dim3(ROWS / 8), blk, 0, stream>>>(ln2_f, y2_f, n3g, n3b,
                                                         (float*)d_out, (unsigned short*)nullptr, ROWS);
}
